// TimeSeriesSDE_10127532884074
// MI455X (gfx1250) — matrix operand order repaired and checked by reading, not compiled, not run on hardware
//
#include <hip/hip_runtime.h>
#include <hip/hip_fp16.h>
#include <math.h>

typedef __attribute__((ext_vector_type(16))) _Float16 v16h;
typedef __attribute__((ext_vector_type(8)))  _Float16 v8h;
typedef __attribute__((ext_vector_type(4)))  _Float16 v4h;
typedef __attribute__((ext_vector_type(8)))  float    v8f;
typedef __attribute__((ext_vector_type(4)))  unsigned int u32x4;
typedef __attribute__((ext_vector_type(8)))  int i32x8;
typedef __attribute__((ext_vector_type(4)))  int i32x4;

#define B_TOT    1024
#define NN       256
#define T_IN     96
#define HID      64
#define TILE_B   16
#define NTHREADS 128
#define H_OUT    10

// ---- dynamic LDS layout (bytes) ----
#define OFF_WT1   0            // 64x256 f16  dw1^T
#define OFF_WT2   32768        // 256x64 f16  dw2^T
#define OFF_VT1   65536        // 64x256 f16  vw1^T
#define OFF_VT2   98304        // 256x64 f16  vw2^T
#define OFF_Y     131072       // 16x256 f32
#define OFF_Y1    147456       // 16x256 f32
#define OFF_F     163840       // 3 x 16x256 f32
#define OFF_G     212992       // 3 x 16x256 f32
#define OFF_HBUF  262144       // 16x256 f16
#define OFF_HID   270336       // 16x64  f16
#define OFF_DW    272384       // 16x256 f32
#define OFF_DZ    288768       // 16x256 f32
#define OFF_BD1   305152       // 64 f32
#define OFF_BD2   305408       // 256 f32
#define OFF_BV1   306432       // 64 f32
#define OFF_BV2   306688       // 256 f32
#define SMEM_BYTES 307712

// Tensor Data Mover availability (5-arg on ROCm7.2/clang-22, 6-arg on clang-23+)
#if defined(__has_builtin)
#if __has_builtin(__builtin_amdgcn_tensor_load_to_lds) && \
    __has_builtin(__builtin_amdgcn_s_wait_tensorcnt)
#define USE_TDM 1
#endif
#endif

struct Sched {
  int n_steps;
  float hs[128];
  int out_slot[128];   // 0 = no output this step, else forecast slot 1..9
};

__device__ __forceinline__ float softplusf(float x) {
  return (x > 20.f) ? x : log1pf(expf(x));
}

#ifdef USE_TDM
// Issue one TDM 2D tile load: 16 rows x 256 f32, contiguous, global -> LDS.
// D# packing per CDNA5 ISA ch.8 (group0 128b, group1 256b; groups 2/3 zero).
__device__ __forceinline__ void tdm_load_tile(unsigned lds_off, const float* gptr) {
  union { unsigned u[4]; u32x4 v; } g0;
  unsigned long long ga = (unsigned long long)(const void*)gptr;
  g0.u[0] = 1u;                                   // count=1 (valid descriptor)
  g0.u[1] = lds_off;                              // lds_addr
  g0.u[2] = (unsigned)(ga & 0xFFFFFFFFu);         // global_addr[31:0]
  g0.u[3] = (unsigned)((ga >> 32) & 0x1FFFFFFu)   // global_addr[56:32]
            | (2u << 30);                         // type=2 ("image")
  union { int i[8]; i32x8 v; } g1;
  g1.i[0] = (int)(2u << 16);          // data_size=2 (4 bytes); mask/flags = 0
  g1.i[1] = (int)(256u << 16);        // tensor_dim0[15:0]=256 (bits 63:48)
  g1.i[2] = (int)(16u << 16);         // tensor_dim1[15:0]=16  (bits 95:80)
  g1.i[3] = (int)(256u << 16);        // tile_dim0=256         (bits 127:112)
  g1.i[4] = 16;                       // tile_dim1=16          (bits 143:128)
  g1.i[5] = 256;                      // tensor_dim0_stride low32 = 256
  g1.i[6] = 0;                        // stride0 hi / stride1 lo
  g1.i[7] = 0;                        // stride1 hi
  i32x4 gz = {0, 0, 0, 0};
#if __clang_major__ >= 23
  i32x8 gz8 = {0, 0, 0, 0, 0, 0, 0, 0};
  __builtin_amdgcn_tensor_load_to_lds(g0.v, g1.v, gz, gz, gz8, 0);
#else
  __builtin_amdgcn_tensor_load_to_lds(g0.v, g1.v, gz, gz, 0);
#endif
}
#endif

// A fragment: 16xK tile, row-major f16 in LDS, lane = M, K layout per ISA
__device__ __forceinline__ v16h afrag(const _Float16* A, int ld, int k0,
                                      int l15, int lhalf) {
  const _Float16* p = A + l15 * ld + k0 + lhalf * 8;
  v8h lo = *(const v8h*)p;          // K = k0+8*lhalf   .. +7
  v8h hi = *(const v8h*)(p + 16);   // K = k0+8*lhalf+16 .. +23
  return __builtin_shufflevector(lo, hi, 0,1,2,3,4,5,6,7,8,9,10,11,12,13,14,15);
}

// B fragment: weights stored transposed Bt[nOut][K]; lane = N, K layout per ISA
__device__ __forceinline__ v16h bfrag(const _Float16* Bt, int ld, int n0, int k0,
                                      int l15, int lhalf) {
  const _Float16* p = Bt + (n0 + l15) * ld + k0 + lhalf * 8;
  v8h lo = *(const v8h*)p;
  v8h hi = *(const v8h*)(p + 16);
  return __builtin_shufflevector(lo, hi, 0,1,2,3,4,5,6,7,8,9,10,11,12,13,14,15);
}

// Two-layer MLP: GEMM1 (16x256 @ 256x64) + act -> hid, GEMM2 (16x64 @ 64x256).
// ACT: 0 = tanh, 1 = softplus (inner activation).
template <int ACT>
__device__ __forceinline__ void mlp(const _Float16* Hbuf, const _Float16* W1t,
                                    const float* b1, const _Float16* W2t,
                                    _Float16* hid, int l15, int lhalf, int wv,
                                    v8f cout[4]) {
  v8f c = {0.f,0.f,0.f,0.f,0.f,0.f,0.f,0.f};
#pragma unroll
  for (int k0 = 0; k0 < NN; k0 += 32) {
    v16h a = afrag(Hbuf, NN, k0, l15, lhalf);
    v16h b = bfrag(W1t, NN, 16 * wv, k0, l15, lhalf);
    c = __builtin_amdgcn_wmma_f32_16x16x32_f16(false, a, false, b, (short)0, c,
                                               false, false);
  }
#pragma unroll
  for (int r = 0; r < 8; ++r) {
    int m = r + 8 * lhalf;
    int nh = 16 * wv + l15;
    float v = c[r] + b1[nh];
    v = ACT ? softplusf(v) : tanhf(v);
    hid[m * HID + nh] = (_Float16)v;
  }
  __syncthreads();
  v16h a0 = afrag(hid, HID, 0, l15, lhalf);
  v16h a1 = afrag(hid, HID, 32, l15, lhalf);
#pragma unroll
  for (int t = 0; t < 4; ++t) {
    int n0 = 64 * wv + 16 * t;
    v8f cc = {0.f,0.f,0.f,0.f,0.f,0.f,0.f,0.f};
    cc = __builtin_amdgcn_wmma_f32_16x16x32_f16(
        false, a0, false, bfrag(W2t, HID, n0, 0, l15, lhalf), (short)0, cc,
        false, false);
    cc = __builtin_amdgcn_wmma_f32_16x16x32_f16(
        false, a1, false, bfrag(W2t, HID, n0, 32, l15, lhalf), (short)0, cc,
        false, false);
    cout[t] = cc;
  }
}

__global__ __launch_bounds__(NTHREADS)
void sde_srk_kernel(const float* __restrict__ x,
                    const float* __restrict__ dw1, const float* __restrict__ db1,
                    const float* __restrict__ dw2, const float* __restrict__ db2,
                    const float* __restrict__ vw1, const float* __restrict__ vb1,
                    const float* __restrict__ vw2, const float* __restrict__ vb2,
                    const float* __restrict__ dW, const float* __restrict__ dZ,
                    float* __restrict__ out, Sched sc) {
  extern __shared__ char smem[];
  _Float16* wT1 = (_Float16*)(smem + OFF_WT1);
  _Float16* wT2 = (_Float16*)(smem + OFF_WT2);
  _Float16* vT1 = (_Float16*)(smem + OFF_VT1);
  _Float16* vT2 = (_Float16*)(smem + OFF_VT2);
  float* yS   = (float*)(smem + OFF_Y);
  float* y1S  = (float*)(smem + OFF_Y1);
  float* fst  = (float*)(smem + OFF_F);
  float* gst  = (float*)(smem + OFF_G);
  _Float16* Hbuf = (_Float16*)(smem + OFF_HBUF);
  _Float16* hid  = (_Float16*)(smem + OFF_HID);
  float* dws = (float*)(smem + OFF_DW);
  float* dzs = (float*)(smem + OFF_DZ);
  float* bd1 = (float*)(smem + OFF_BD1);
  float* bd2 = (float*)(smem + OFF_BD2);
  float* bv1 = (float*)(smem + OFF_BV1);
  float* bv2 = (float*)(smem + OFF_BV2);

  const int tid   = threadIdx.x;
  const int lane  = tid & 31;
  const int wv    = tid >> 5;
  const int l15   = lane & 15;
  const int lhalf = lane >> 4;
  const int samp  = blockIdx.x >> 6;
  const int tile  = blockIdx.x & 63;
  const int b0    = tile * TILE_B;

  // Butcher / beta tables (SRI2)
  const float A0c[4][3] = {{0,0,0},{1.f,0,0},{0.25f,0.25f,0},{0,0,0}};
  const float A1c[4][3] = {{0,0,0},{0.25f,0,0},{1.f,0,0},{0,0,0.25f}};
  const float B0c[4][3] = {{0,0,0},{0,0,0},{1.f,0.5f,0},{0,0,0}};
  const float B1c[4][3] = {{0,0,0},{-0.5f,0,0},{1.f,0,0},{2.f,-1.f,0.5f}};
  const float ALPHAc[4] = {1.f/6.f, 1.f/6.f, 2.f/3.f, 0.f};
  const float BT1[4] = {-1.f, 4.f/3.f, 2.f/3.f, 0.f};
  const float BT2[4] = {1.f, -4.f/3.f, 1.f/3.f, 0.f};
  const float BT3[4] = {2.f, -4.f/3.f, -2.f/3.f, 0.f};
  const float BT4[4] = {-2.f, 5.f/3.f, -2.f/3.f, 1.f};
  const float INV_SQRT3 = 0.5773502691896258f;

  // ---- load weights transposed to f16 in LDS ----
  for (int i = tid; i < NN * HID; i += NTHREADS) {
    int n = i / HID, h = i % HID;          // dw1/vw1: (256,64)
    wT1[h * NN + n] = (_Float16)dw1[i];
    vT1[h * NN + n] = (_Float16)vw1[i];
  }
  for (int i = tid; i < HID * NN; i += NTHREADS) {
    int h = i / NN, n = i % NN;            // dw2/vw2: (64,256)
    wT2[n * HID + h] = (_Float16)dw2[i];
    vT2[n * HID + h] = (_Float16)vw2[i];
  }
  for (int i = tid; i < HID; i += NTHREADS) { bd1[i] = db1[i]; bv1[i] = vb1[i]; }
  for (int i = tid; i < NN;  i += NTHREADS) { bd2[i] = db2[i]; bv2[i] = vb2[i]; }

  // ---- init y = x[:, :, -1]; write forecast slot 0 ----
#pragma unroll
  for (int i = 0; i < 8; ++i) {
    int e = (tid + i * NTHREADS) * 4;
    int m = e >> 8, n = e & 255;
    const float* xr = x + ((size_t)(b0 + m) * NN + n) * T_IN + (T_IN - 1);
    float4 v;
    v.x = xr[0]; v.y = xr[T_IN]; v.z = xr[2 * T_IN]; v.w = xr[3 * T_IN];
    *(float4*)(yS + e) = v;
    *(float4*)(y1S + e) = v;
    size_t ob = ((size_t)(samp * B_TOT + b0 + m) * NN + n) * H_OUT;
    out[ob] = v.x; out[ob + H_OUT] = v.y; out[ob + 2*H_OUT] = v.z; out[ob + 3*H_OUT] = v.w;
  }
  __syncthreads();

  const int nsteps = sc.n_steps;
  for (int step = 0; step < nsteps; ++step) {
    const float h = sc.hs[step];
    const float sqrt_h = sqrtf(h);
    const float inv_sqrt_h = 1.f / sqrt_h;
    const float inv_6h = 1.f / (6.f * h);

    const size_t nbase = ((size_t)(samp * nsteps + step) * B_TOT + b0) * NN;
#ifdef USE_TDM
    // ---- async: Tensor Data Mover streams this step's noise tiles to LDS.
    // First consumer is the stage-0 diffusion update, ~6 compute phases away,
    // so the DMA overlaps with WMMA work. Wave 0 issues; waits before use.
    if (wv == 0) {
      tdm_load_tile(OFF_DW, dW + nbase);
      tdm_load_tile(OFF_DZ, dZ + nbase);
    }
#else
    // ---- fallback: vector loads of the noise tile ----
#pragma unroll
    for (int i = 0; i < 8; ++i) {
      int e = (tid + i * NTHREADS) * 4;
      *(float4*)(dws + e) = *(const float4*)(dW + nbase + e);
      *(float4*)(dzs + e) = *(const float4*)(dZ + nbase + e);
    }
    __syncthreads();
#endif
    if (step + 1 < nsteps) {
      const size_t nnext = nbase + (size_t)B_TOT * NN;
      __builtin_prefetch(dW + nnext + tid * 32, 0, 0);  // global_prefetch_b8
      __builtin_prefetch(dZ + nnext + tid * 32, 0, 0);
    }

    for (int stage = 0; stage < 4; ++stage) {
      if (stage < 3) {
        // ---- H0 combine -> f16 Hbuf (reads noise only for stage >= 1) ----
#pragma unroll
        for (int i = 0; i < 8; ++i) {
          int e = (tid + i * NTHREADS) * 4;
          const float* yp = yS + e;
          const float* dp = dws + e;
          const float* zp = dzs + e;
          float acc[4];
#pragma unroll
          for (int k = 0; k < 4; ++k) acc[k] = yp[k];
          for (int j = 0; j < stage; ++j) {
            float ca = A0c[stage][j] * h;
            float cb = B0c[stage][j];
            const float* fp = fst + j * (TILE_B * NN) + e;
            const float* gp = gst + j * (TILE_B * NN) + e;
#pragma unroll
            for (int k = 0; k < 4; ++k) {
              float ik0h = 0.5f * sqrt_h * (dp[k] + zp[k] * INV_SQRT3); // I_k0/h
              acc[k] += ca * fp[k] + cb * ik0h * gp[k];
            }
          }
          v4h hv;
#pragma unroll
          for (int k = 0; k < 4; ++k) hv[k] = (_Float16)acc[k];
          *(v4h*)(Hbuf + e) = hv;
        }
        __syncthreads();

        // ---- drift MLP + store f_s, accumulate y1 ----
        v8f cf[4];
        mlp<0>(Hbuf, wT1, bd1, wT2, hid, l15, lhalf, wv, cf);
        const float ah = ALPHAc[stage] * h;
#pragma unroll
        for (int t = 0; t < 4; ++t) {
#pragma unroll
          for (int r = 0; r < 8; ++r) {
            int m = r + 8 * lhalf;
            int ng = 64 * wv + 16 * t + l15;
            float v = cf[t][r] + bd2[ng];
            int idx = m * NN + ng;
            fst[stage * (TILE_B * NN) + idx] = v;
            y1S[idx] += ah * v;
          }
        }
        __syncthreads();
      }

      // ---- H1 combine -> f16 Hbuf ----
#pragma unroll
      for (int i = 0; i < 8; ++i) {
        int e = (tid + i * NTHREADS) * 4;
        const float* yp = yS + e;
        float acc[4];
#pragma unroll
        for (int k = 0; k < 4; ++k) acc[k] = yp[k];
        for (int j = 0; j < stage; ++j) {
          float ca = A1c[stage][j] * h;
          float cb = B1c[stage][j] * sqrt_h;
          const float* fp = fst + j * (TILE_B * NN) + e;
          const float* gp = gst + j * (TILE_B * NN) + e;
#pragma unroll
          for (int k = 0; k < 4; ++k) acc[k] += ca * fp[k] + cb * gp[k];
        }
        v4h hv;
#pragma unroll
        for (int k = 0; k < 4; ++k) hv[k] = (_Float16)acc[k];
        *(v4h*)(Hbuf + e) = hv;
      }
      __syncthreads();

      // ---- diffusion MLP ----
      v8f cg[4];
      mlp<1>(Hbuf, vT1, bv1, vT2, hid, l15, lhalf, wv, cg);

#ifdef USE_TDM
      if (stage == 0) {
        // First use of the noise tile: drain the TDM and publish to all waves.
        if (wv == 0) __builtin_amdgcn_s_wait_tensorcnt(0);
        __syncthreads();
      }
#endif

      // ---- store g_s, accumulate y1 with stochastic weights ----
#pragma unroll
      for (int t = 0; t < 4; ++t) {
#pragma unroll
        for (int r = 0; r < 8; ++r) {
          int m = r + 8 * lhalf;
          int ng = 64 * wv + 16 * t + l15;
          int idx = m * NN + ng;
          float gval = softplusf(cg[t][r] + bv2[ng]);
          float d = dws[idx], z = dzs[idx];
          float Ik = sqrt_h * d;
          float Ik0h = 0.5f * sqrt_h * (d + z * INV_SQRT3);      // I_k0 / h
          float Ikk = 0.5f * (Ik * Ik - h);
          float Ikkk_h = (Ik * Ik * Ik - 3.f * h * Ik) * inv_6h; // I_kkk / h
          float gw = BT1[stage] * Ik + BT2[stage] * Ikk * inv_sqrt_h +
                     BT3[stage] * Ik0h + BT4[stage] * Ikkk_h;
          y1S[idx] += gw * gval;
          if (stage < 3) gst[stage * (TILE_B * NN) + idx] = gval;
        }
      }
      __syncthreads();
    }

    // ---- commit y <- y1, emit forecast slot if scheduled ----
    const int slot = sc.out_slot[step];
#pragma unroll
    for (int i = 0; i < 8; ++i) {
      int e = (tid + i * NTHREADS) * 4;
      float4 v = *(float4*)(y1S + e);
      *(float4*)(yS + e) = v;
      if (slot) {
        int m = e >> 8, n = e & 255;
        size_t ob = ((size_t)(samp * B_TOT + b0 + m) * NN + n) * H_OUT + slot;
        out[ob] = v.x; out[ob + H_OUT] = v.y; out[ob + 2*H_OUT] = v.z; out[ob + 3*H_OUT] = v.w;
      }
    }
    __syncthreads();
  }
}

extern "C" void kernel_launch(void* const* d_in, const int* in_sizes, int n_in,
                              void* d_out, int out_size, void* d_ws, size_t ws_size,
                              hipStream_t stream) {
  (void)in_sizes; (void)n_in; (void)out_size; (void)d_ws; (void)ws_size;

  // Build the integration schedule (mirrors reference _schedule()).
  Sched sc;
  {
    double t = 0.0;
    int n = 0;
    for (int i = 1; i < 10; ++i) {
      double nt = 10.0 * (double)i / 9.0;
      while (t < nt - 1e-9) {
        double hh = 0.1;
        if (nt - t < hh) hh = nt - t;
        sc.hs[n] = (float)hh;
        sc.out_slot[n] = 0;
        ++n;
        t += hh;
      }
      t = nt;
      sc.out_slot[n - 1] = i;
    }
    sc.n_steps = n;
    for (int i = n; i < 128; ++i) { sc.hs[i] = 0.f; sc.out_slot[i] = 0; }
  }

  const float* x   = (const float*)d_in[0];
  const float* dw1 = (const float*)d_in[1];
  const float* db1 = (const float*)d_in[2];
  const float* dw2 = (const float*)d_in[3];
  const float* db2 = (const float*)d_in[4];
  const float* vw1 = (const float*)d_in[5];
  const float* vb1 = (const float*)d_in[6];
  const float* vw2 = (const float*)d_in[7];
  const float* vb2 = (const float*)d_in[8];
  const float* dWp = (const float*)d_in[9];
  const float* dZp = (const float*)d_in[10];
  float* out = (float*)d_out;

  (void)hipFuncSetAttribute(reinterpret_cast<const void*>(&sde_srk_kernel),
                            hipFuncAttributeMaxDynamicSharedMemorySize,
                            (int)SMEM_BYTES);

  dim3 grid(2 * (B_TOT / TILE_B));  // 128 workgroups: 2 samples x 64 row tiles
  dim3 block(NTHREADS);
  sde_srk_kernel<<<grid, block, SMEM_BYTES, stream>>>(
      x, dw1, db1, dw2, db2, vw1, vb1, vw2, vb2, dWp, dZp, out, sc);
}
